// MultiSequenceEventTokenizer_15410342658357
// MI455X (gfx1250) — matrix operand order repaired and checked by reading, not compiled, not run on hardware
//
#include <hip/hip_runtime.h>
#include <hip/hip_bf16.h>

// ---------------------------------------------------------------------------
// MultiSequenceEventTokenizer for MI455X (gfx1250, wave32, WMMA)
//
// Pipeline:
//   K1 build_xn  : 5-way embedding gather + LayerNorm -> bf16 xn [32768,2560]
//   K2 transpose_cvt : w1,w2 fp32 [K][N] -> bf16 TRANSPOSED [N][K] (workspace)
//   K3 gemm bf16 WMMA : mid = SiLU(xn@w1+b1)  (bf16 out)  [32768,2048]
//   K4 gemm bf16 WMMA : event = mid@w2+b2     (fp32 out)  [32768,512]
//   K5 zero d_out ; K6 regroup scatter -> states[B,S,M,H], smask[B,S,M]
//
// GEMM: 64(M)x128(N) block tile, BK=32, 4 waves; wave strip 16x128 -> 8x
// v_wmma_f32_16x16x32_bf16 per K-step. Double-buffered LDS with register
// staging: next tile's global_load_b128s issue before compute so they overlap
// the WMMAs; every LDS access (both staging and fragment build) is b128.
// Fragment lane layouts per ISA 7.12.2 (wave32):
//   A 16x32 bf16: lane<16 row=lane, halves = K[akhi..+7],[akhi+16..+23]
//   B 32x16 bf16: lane holds one column, halves = K[bkhi..+7],[bkhi+16..+23] (same as A)
//     (weights pre-transposed to [N][K], so LDS tile rows are K-contiguous)
// ---------------------------------------------------------------------------

#define B_  16
#define L_  2048
#define H_  512
#define S_  8
#define M_  512
#define TG_ 64
#define XDIM (5 * H_)      // 2560
#define MIDN (4 * H_)      // 2048
#define ROWS (B_ * L_)     // 32768
#define LN_EPS 1e-5f

typedef unsigned short ushort;
typedef __attribute__((ext_vector_type(16))) __bf16  v16bf;
typedef __attribute__((ext_vector_type(8)))  float   v8f;
typedef __attribute__((ext_vector_type(8)))  ushort  v8us;   // 16 bytes

__device__ __forceinline__ ushort f32_to_bf16(float f) {
    unsigned int u = __float_as_uint(f);
    unsigned int r = u + 0x7FFFu + ((u >> 16) & 1u);   // round-to-nearest-even
    return (ushort)(r >> 16);
}

// ---------------------------------------------------------------------------
// K1: gather 4 shared-vocab embeddings + time-gap embedding, LayerNorm(2560),
//     write bf16 normalized row. One block (256 thr) per event row.
// ---------------------------------------------------------------------------
__global__ void __launch_bounds__(256)
build_xn_kernel(const int* __restrict__ tok, const int* __restrict__ ptok,
                const int* __restrict__ atok, const int* __restrict__ ctok,
                const int* __restrict__ tgap,
                const float* __restrict__ emb, const float* __restrict__ gapt,
                const float* __restrict__ gamma, const float* __restrict__ beta,
                ushort* __restrict__ Xn) {
    const int row = blockIdx.x;
    const int tid = threadIdx.x;

    const int t0 = tok[row], t1 = ptok[row], t2 = atok[row], t3 = ctok[row];
    int g = tgap[row];
    g = g < 0 ? 0 : (g > TG_ ? TG_ : g);

    float v[10];
    float sum = 0.f, sq = 0.f;
#pragma unroll
    for (int e = 0; e < 10; ++e) {
        const int j   = e * 256 + tid;     // 0..2559
        const int seg = j >> 9;            // /512
        const int off = j & 511;
        float val;
        if      (seg == 0) val = emb[(size_t)t0 * H_ + off];
        else if (seg == 1) val = emb[(size_t)t1 * H_ + off];
        else if (seg == 2) val = emb[(size_t)t2 * H_ + off];
        else if (seg == 3) val = emb[(size_t)t3 * H_ + off];
        else               val = (g != 0) ? gapt[(size_t)g * H_ + off] : 0.f;
        v[e] = val;
        sum += val;
        sq  += val * val;
    }

    __shared__ float ssum[256];
    __shared__ float ssq[256];
    ssum[tid] = sum; ssq[tid] = sq;
    __syncthreads();
    for (int off = 128; off > 0; off >>= 1) {
        if (tid < off) { ssum[tid] += ssum[tid + off]; ssq[tid] += ssq[tid + off]; }
        __syncthreads();
    }
    const float inv_n = 1.f / (float)XDIM;
    const float mu  = ssum[0] * inv_n;
    const float var = ssq[0] * inv_n - mu * mu;
    const float rs  = rsqrtf(var + LN_EPS);

    ushort* __restrict__ out = Xn + (size_t)row * XDIM;
#pragma unroll
    for (int e = 0; e < 10; ++e) {
        const int j = e * 256 + tid;
        const float xn = (v[e] - mu) * rs * gamma[j] + beta[j];
        out[j] = f32_to_bf16(xn);
    }
}

// ---------------------------------------------------------------------------
// K2: fp32 [K][N] -> bf16 transposed [N][K], LDS-tiled 32x32 for coalescing.
// block (32,8); grid (N/32, K/32).
// ---------------------------------------------------------------------------
__global__ void __launch_bounds__(256)
transpose_cvt_kernel(const float* __restrict__ src, ushort* __restrict__ dst,
                     int K, int N) {
    __shared__ float t[32][33];
    const int n0 = blockIdx.x * 32;
    const int k0 = blockIdx.y * 32;
    const int tx = threadIdx.x;
    const int ty = threadIdx.y;
#pragma unroll
    for (int yy = 0; yy < 32; yy += 8)
        t[ty + yy][tx] = src[(size_t)(k0 + ty + yy) * N + (n0 + tx)];
    __syncthreads();
#pragma unroll
    for (int yy = 0; yy < 32; yy += 8)
        dst[(size_t)(n0 + ty + yy) * K + (k0 + tx)] = f32_to_bf16(t[tx][ty + yy]);
}

// ---------------------------------------------------------------------------
// K3/K4: bf16 WMMA GEMM. A row-major [M][K] bf16; W transposed [N][K] bf16.
// Block tile 64(M)x128(N), BK=32, 128 threads (4 waves), double-buffered LDS.
// ---------------------------------------------------------------------------
template <int SILU, int OUTBF16>
__global__ void __launch_bounds__(128)
gemm_bf16_wmma(const ushort* __restrict__ A, const ushort* __restrict__ Wt,
               const float* __restrict__ bias, void* __restrict__ outp,
               int Mdim, int Ndim, int Kdim) {
    constexpr int BM = 64, BN = 128, BK = 32;
    constexpr int LDT = BK + 8;                        // 40 ushorts = 80 B
    __shared__ __align__(16) ushort As[2][BM][LDT];    // 10.0 KB
    __shared__ __align__(16) ushort Bs[2][BN][LDT];    // 20.0 KB

    const int tid  = threadIdx.x;
    const int wave = tid >> 5;
    const int lane = tid & 31;
    const int m0 = blockIdx.y * BM;
    const int n0 = blockIdx.x * BN;

    v8f acc[8];
#pragma unroll
    for (int s = 0; s < 8; ++s)
#pragma unroll
        for (int r = 0; r < 8; ++r) acc[s][r] = 0.f;

    const int arow = wave * 16 + (lane & 15);
    const int akhi = (lane & 16) ? 8 : 0;          // A: K-half select
    const int bkhi = (lane & 16) ? 8 : 0;          // B: K-half select
    const int bcol = lane & 15;

    // per-thread staging chunk coordinates (16B chunks, 4 chunks per tile row)
    const int ar0 = tid >> 2,          ac0 = (tid & 3) * 8;          // A chunk 0
    const int ar1 = (tid + 128) >> 2,  ac1 = (tid & 3) * 8;          // A chunk 1
    union Frag { v8us h[2]; v16bf v; };
    v8us ra[2], rb[4];

    // ---- prologue: load + store tile 0 ----
    ra[0] = *(const v8us*)&A[(size_t)(m0 + ar0) * Kdim + ac0];
    ra[1] = *(const v8us*)&A[(size_t)(m0 + ar1) * Kdim + ac1];
#pragma unroll
    for (int c = 0; c < 4; ++c) {
        const int i = tid + c * 128;
        rb[c] = *(const v8us*)&Wt[(size_t)(n0 + (i >> 2)) * Kdim + (i & 3) * 8];
    }
    *(v8us*)&As[0][ar0][ac0] = ra[0];
    *(v8us*)&As[0][ar1][ac1] = ra[1];
#pragma unroll
    for (int c = 0; c < 4; ++c) {
        const int i = tid + c * 128;
        *(v8us*)&Bs[0][i >> 2][(i & 3) * 8] = rb[c];
    }
    __syncthreads();

    int buf = 0;
    for (int k0 = 0; k0 < Kdim; k0 += BK) {
        const bool hasNext = (k0 + BK) < Kdim;
        // ---- issue next tile's global loads (overlap with WMMA below) ----
        if (hasNext) {
            const int kn = k0 + BK;
            ra[0] = *(const v8us*)&A[(size_t)(m0 + ar0) * Kdim + (kn + ac0)];
            ra[1] = *(const v8us*)&A[(size_t)(m0 + ar1) * Kdim + (kn + ac1)];
#pragma unroll
            for (int c = 0; c < 4; ++c) {
                const int i = tid + c * 128;
                rb[c] = *(const v8us*)&Wt[(size_t)(n0 + (i >> 2)) * Kdim +
                                          (kn + (i & 3) * 8)];
            }
        }

        // ---- compute: A fragment once, 8 N-subtiles ----
        Frag af;
        af.h[0] = *(const v8us*)&As[buf][arow][akhi];        // K akhi..+7
        af.h[1] = *(const v8us*)&As[buf][arow][akhi + 16];   // K akhi+16..+23
#pragma unroll
        for (int s = 0; s < 8; ++s) {
            const ushort* bp = &Bs[buf][s * 16 + bcol][bkhi];
            Frag bf;
            bf.h[0] = *(const v8us*)bp;                      // K bkhi..+7
            bf.h[1] = *(const v8us*)(bp + 16);               // K bkhi+16..+23
            acc[s] = __builtin_amdgcn_wmma_f32_16x16x32_bf16(
                false, af.v, false, bf.v, (short)0, acc[s], false, false);
        }

        // ---- store staged registers into the other buffer ----
        if (hasNext) {
            const int nb = buf ^ 1;
            *(v8us*)&As[nb][ar0][ac0] = ra[0];
            *(v8us*)&As[nb][ar1][ac1] = ra[1];
#pragma unroll
            for (int c = 0; c < 4; ++c) {
                const int i = tid + c * 128;
                *(v8us*)&Bs[nb][i >> 2][(i & 3) * 8] = rb[c];
            }
        }
        __syncthreads();
        buf ^= 1;
    }

    // ---- epilogue: bias (+SiLU), store ----
    const int rbase = m0 + wave * 16 + ((lane & 16) ? 8 : 0);
    ushort* outb = (ushort*)outp;
    float*  outf = (float*)outp;
#pragma unroll
    for (int s = 0; s < 8; ++s) {
        const int col = n0 + s * 16 + bcol;
        const float bv = bias[col];
#pragma unroll
        for (int r = 0; r < 8; ++r) {
            const int row = rbase + r;
            float val = acc[s][r] + bv;
            if (SILU) val = val / (1.f + __expf(-val));   // x*sigmoid(x)
            if (OUTBF16) outb[(size_t)row * Ndim + col] = f32_to_bf16(val);
            else         outf[(size_t)row * Ndim + col] = val;
        }
    }
}

// ---------------------------------------------------------------------------
// K5: zero output (states + smask regions)
// ---------------------------------------------------------------------------
__global__ void zero_f32_kernel(float* __restrict__ p, size_t n) {
    for (size_t i = blockIdx.x * (size_t)blockDim.x + threadIdx.x; i < n;
         i += (size_t)gridDim.x * blockDim.x)
        p[i] = 0.f;
}

// ---------------------------------------------------------------------------
// K6: ragged regroup. One block (256 thr) per (b,s). Count matches, keep last
// M, scatter event rows (+pos+sid) into states, set smask. Empty groups get
// empty_tokens at slot 0.
// ---------------------------------------------------------------------------
__global__ void __launch_bounds__(256)
regroup_kernel(const float* __restrict__ event, const float* __restrict__ pos,
               const float* __restrict__ sidt, const float* __restrict__ emptyt,
               const int* __restrict__ gid, const int* __restrict__ maskp,
               float* __restrict__ states, float* __restrict__ smask) {
    const int bs = blockIdx.x;
    const int b = bs / S_;
    const int s = bs % S_;
    const int tid = threadIdx.x;
    const int want = s + 1;

    const int* grow = gid + (size_t)b * L_;
    const int* mrow = maskp + (size_t)b * L_;
    const float* sid = sidt + (size_t)want * H_;

    __shared__ int sc[256];
    __shared__ int kl[256];
    __shared__ int ksl[256];
    __shared__ int kcnt;

    // pass 1: total matches for this (b,s)
    int cl = 0;
    for (int l = tid; l < L_; l += 256)
        cl += (mrow[l] != 0 && grow[l] == want) ? 1 : 0;
    sc[tid] = cl;
    __syncthreads();
    for (int off = 128; off > 0; off >>= 1) {
        if (tid < off) sc[tid] += sc[tid + off];
        __syncthreads();
    }
    const int count = sc[0];
    __syncthreads();
    const int offset = count > M_ ? count - M_ : 0;

    if (count == 0) {
        // learned empty token at slot 0
        for (int j = tid; j < H_; j += 256)
            states[((size_t)bs * M_) * H_ + j] = emptyt[(size_t)s * H_ + j] + pos[j] + sid[j];
        if (tid == 0) smask[(size_t)bs * M_] = 1.0f;
        return;
    }

    // pass 2: tiled inclusive scan over L for ranks, compact kept entries
    int base = 0;
    for (int t0 = 0; t0 < L_; t0 += 256) {
        const int l = t0 + tid;
        const int mv = (l < L_ && mrow[l] != 0 && grow[l] == want) ? 1 : 0;
        sc[tid] = mv;
        __syncthreads();
        for (int off = 1; off < 256; off <<= 1) {
            int v = (tid >= off) ? sc[tid - off] : 0;
            __syncthreads();
            sc[tid] += v;
            __syncthreads();
        }
        const int incl = sc[tid];
        const int tile_total = sc[255];
        if (tid == 0) kcnt = 0;
        __syncthreads();
        if (mv) {
            const int slot = base + incl - 1 - offset;
            if (slot >= 0 && slot < M_) {
                const int p = atomicAdd(&kcnt, 1);   // distinct slots; order free
                kl[p] = l;
                ksl[p] = slot;
            }
        }
        __syncthreads();
        const int nk = kcnt;
        for (int e = 0; e < nk; ++e) {
            const int le = kl[e];
            const int sl = ksl[e];
            const float* ev = event + ((size_t)b * L_ + le) * H_;
            float* dst = states + ((size_t)bs * M_ + sl) * H_;
            const float* pr = pos + (size_t)sl * H_;
            for (int j = tid; j < H_; j += 256) dst[j] = ev[j] + pr[j] + sid[j];
            if (tid == 0) smask[(size_t)bs * M_ + sl] = 1.0f;
        }
        base += tile_total;
        __syncthreads();
    }
}

// ---------------------------------------------------------------------------
// launch
// ---------------------------------------------------------------------------
extern "C" void kernel_launch(void* const* d_in, const int* in_sizes, int n_in,
                              void* d_out, int out_size, void* d_ws, size_t ws_size,
                              hipStream_t stream) {
    (void)in_sizes; (void)n_in; (void)out_size; (void)ws_size;

    const int*   tok   = (const int*)  d_in[0];
    const int*   ptok  = (const int*)  d_in[1];
    const int*   atok  = (const int*)  d_in[2];
    const int*   ctok  = (const int*)  d_in[3];
    const int*   tgap  = (const int*)  d_in[4];
    const int*   gid   = (const int*)  d_in[5];
    const int*   mask  = (const int*)  d_in[6];
    const float* emb   = (const float*)d_in[7];
    const float* gapt  = (const float*)d_in[8];
    const float* sidt  = (const float*)d_in[9];
    const float* post  = (const float*)d_in[10];
    const float* gamma = (const float*)d_in[11];
    const float* beta  = (const float*)d_in[12];
    const float* w1    = (const float*)d_in[13];
    const float* b1    = (const float*)d_in[14];
    const float* w2    = (const float*)d_in[15];
    const float* b2    = (const float*)d_in[16];
    const float* emptyt= (const float*)d_in[17];

    // workspace: Xn bf16 | W1t bf16 [2048][2560] | W2t bf16 [512][2048] |
    //            Mid bf16 | Event f32
    ushort* Xn  = (ushort*)d_ws;                         // ROWS*XDIM
    ushort* W1t = Xn  + (size_t)ROWS * XDIM;             // MIDN*XDIM (transposed)
    ushort* W2t = W1t + (size_t)MIDN * XDIM;             // H_*MIDN  (transposed)
    ushort* Mid = W2t + (size_t)H_ * MIDN;               // ROWS*MIDN
    float*  Evt = (float*)(Mid + (size_t)ROWS * MIDN);   // ROWS*H_

    float* states = (float*)d_out;                                   // B*S*M*H
    float* smask  = states + (size_t)B_ * S_ * M_ * H_;              // B*S*M

    // K1: gather + LayerNorm -> bf16
    build_xn_kernel<<<ROWS, 256, 0, stream>>>(tok, ptok, atok, ctok, tgap,
                                              emb, gapt, gamma, beta, Xn);
    // K2: weight convert + transpose (w1: [2560][2048] -> [2048][2560])
    {
        dim3 blk(32, 8);
        dim3 g1(MIDN / 32, XDIM / 32);
        transpose_cvt_kernel<<<g1, blk, 0, stream>>>(w1, W1t, XDIM, MIDN);
        dim3 g2(H_ / 32, MIDN / 32);
        transpose_cvt_kernel<<<g2, blk, 0, stream>>>(w2, W2t, MIDN, H_);
    }

    // K3: mid = SiLU(xn @ w1 + b1), bf16 out. M=32768 N=2048 K=2560
    {
        dim3 grid(MIDN / 128, ROWS / 64);
        gemm_bf16_wmma<1, 1><<<grid, 128, 0, stream>>>(Xn, W1t, b1, (void*)Mid,
                                                       ROWS, MIDN, XDIM);
    }
    // K4: event = mid @ w2 + b2, fp32 out. M=32768 N=512 K=2048
    {
        dim3 grid(H_ / 128, ROWS / 64);
        gemm_bf16_wmma<0, 0><<<grid, 128, 0, stream>>>(Mid, W2t, b2, (void*)Evt,
                                                       ROWS, H_, MIDN);
    }

    // K5: zero states + smask
    zero_f32_kernel<<<2048, 256, 0, stream>>>(
        states, (size_t)B_ * S_ * M_ * H_ + (size_t)B_ * S_ * M_);

    // K6: ragged regroup scatter
    regroup_kernel<<<B_ * S_, 256, 0, stream>>>(Evt, post, sidt, emptyt,
                                                gid, mask, states, smask);
}
